// GNNPolicy_class_58884001628287
// MI455X (gfx1250) — matrix operand order repaired and checked by reading, not compiled, not run on hardware
//
#include <hip/hip_runtime.h>
#include <hip/hip_bf16.h>

// ---------------------------------------------------------------------------
// CDNA5 (gfx1250) fused GNN-policy forward.
// wave32; WMMA f32_16x16x32_f16 for all 64-wide GEMMs.
// ---------------------------------------------------------------------------

typedef _Float16 v16h __attribute__((ext_vector_type(16)));
typedef _Float16 v8h  __attribute__((ext_vector_type(8)));
typedef float    v8f  __attribute__((ext_vector_type(8)));

#define ENC_NEG_INF 0x007FFFFFu   // enc(-inf): order-preserving uint encoding

__device__ __forceinline__ unsigned enc_f32(float x){
  unsigned u = __float_as_uint(x);
  return (u & 0x80000000u) ? ~u : (u | 0x80000000u);
}
__device__ __forceinline__ float dec_f32(unsigned e){
  unsigned u = (e & 0x80000000u) ? (e & 0x7fffffffu) : ~e;
  return __uint_as_float(u);
}
__device__ __forceinline__ v8f fzero8(){
  v8f r;
#pragma unroll
  for (int i = 0; i < 8; ++i) r[i] = 0.f;
  return r;
}

// Pack a 64x64 row-major f32 weight W[k][n] into 8 WMMA B-fragments (f16).
// Fragment index = colTile(t:0..3)*2 + kChunk(c:0..1). Each fragment:
// 32 lanes x 16 halves; lane holds col n = t*16 + (lane&15),
// halves j = K rows (c*32 + 16*(lane>=16) + j). Requires 256 threads.
__device__ __forceinline__ void pack_frags_64x64(const float* W, _Float16* dst){
  int tid  = threadIdx.x;
  int frag = tid >> 5, lane = tid & 31;
  int t = frag >> 1, c = frag & 1;
  int n  = t * 16 + (lane & 15);
  int kb = c * 32 + ((lane & 16) ? 16 : 0);
  _Float16* d = dst + frag * 512 + lane * 16;
#pragma unroll
  for (int j = 0; j < 16; ++j) d[j] = (_Float16)W[(kb + j) * 64 + n];
}

// 16x64 GEMM: act (16x64 f16, row-major in LDS) x W (64x64 packed frags)
// -> acc[4] 16x16 f32 tiles (D layout: lane n=lane&15, vgpr r -> m = r+8*(lane>=16)).
__device__ __forceinline__ void gemm_tile(const _Float16* act, const _Float16* fragW,
                                          int lane, v8f acc[4]){
  int m  = lane & 15;
  int hi = (lane >> 4) & 1;
  // A-layout (matches packed B): lane holds row m; halves j map to
  // K = chunk*32 + 16*hi + j  => one contiguous 16-half block per frag.
  const v8h* rp = (const v8h*)(act + m * 64);
  v8h l0 = rp[2 * hi],     h0 = rp[2 * hi + 1];
  v8h l1 = rp[4 + 2 * hi], h1 = rp[5 + 2 * hi];
  v16h a0 = __builtin_shufflevector(l0, h0, 0,1,2,3,4,5,6,7,8,9,10,11,12,13,14,15);
  v16h a1 = __builtin_shufflevector(l1, h1, 0,1,2,3,4,5,6,7,8,9,10,11,12,13,14,15);
  const v16h* bp = (const v16h*)fragW;
#pragma unroll
  for (int t = 0; t < 4; ++t){
    v16h b0 = bp[(t * 2 + 0) * 32 + lane];
    v16h b1 = bp[(t * 2 + 1) * 32 + lane];
    acc[t] = __builtin_amdgcn_wmma_f32_16x16x32_f16(false, a0, false, b0,
                                                    (short)0, acc[t], false, false);
    acc[t] = __builtin_amdgcn_wmma_f32_16x16x32_f16(false, a1, false, b1,
                                                    (short)0, acc[t], false, false);
  }
}

// Stage 128 rows: coalesced load, row LayerNorm, f16 tile into LDS.
// 256 threads: thread t handles row t/2, 32 cols (t&1)*32.
__device__ __forceinline__ void stage_ln(const float* __restrict__ X, int N, int base,
                                         const float* __restrict__ lnw,
                                         const float* __restrict__ lnb,
                                         _Float16* sAct, float* sRed, float* sRed2){
  int tid = threadIdx.x;
  int r = tid >> 1, hf = tid & 1;
  int row = base + r;
  float x[32];
  if (row < N){
    const float4* src = (const float4*)(X + (size_t)row * 64 + hf * 32);
#pragma unroll
    for (int i = 0; i < 8; ++i){
      float4 f = src[i];
      x[4*i] = f.x; x[4*i+1] = f.y; x[4*i+2] = f.z; x[4*i+3] = f.w;
    }
  } else {
#pragma unroll
    for (int i = 0; i < 32; ++i) x[i] = 0.f;
  }
  float s = 0.f, ss = 0.f;
#pragma unroll
  for (int i = 0; i < 32; ++i){ s += x[i]; ss += x[i] * x[i]; }
  sRed[tid] = s; sRed2[tid] = ss;
  __syncthreads();
  float mean = (sRed[2*r] + sRed[2*r+1]) * (1.f / 64.f);
  float var  = (sRed2[2*r] + sRed2[2*r+1]) * (1.f / 64.f) - mean * mean;
  float rstd = rsqrtf(var + 1e-5f);
#pragma unroll
  for (int i = 0; i < 32; ++i){
    int col = hf * 32 + i;
    sAct[r * 64 + col] = (_Float16)((x[i] - mean) * rstd * lnw[col] + lnb[col]);
  }
  __syncthreads();
}

// ---------------------------------------------------------------------------
// Kernel 0: tiny front-end (24x768 MLP, 24-token MHA, LN, q-projections),
// weight-fragment packing, accumulator init. One block of 256 threads.
// ---------------------------------------------------------------------------
struct K0Args {
  const float *vsem, *csem, *w1, *b1, *w2, *b2;
  const float *sa_wq, *sa_bq, *sa_wk, *sa_bk, *sa_wv, *sa_bv, *sa_wo, *sa_bo;
  const float *lnw, *lnb;
  const float *cav_wq, *cav_bq, *cac_wq, *cac_bq;
  const float *sv_w, *sc_w, *cav_wk, *cav_wv, *cac_wk, *cac_wv;
  float *qv, *qc;
  unsigned *maxv, *maxc;
  float *numv, *zv, *numc, *zc;
  _Float16 *frags;
};

__global__ __launch_bounds__(256) void k_setup(K0Args A){
  __shared__ float sh1[24 * 128];
  __shared__ float sf [24 * 64];
  __shared__ float sq [24 * 64];
  __shared__ float sk [24 * 64];
  __shared__ float sv [24 * 64];
  __shared__ float so [24 * 64];
  int tid = threadIdx.x;

  // h1 = relu(fea @ proj_w1 + b1), fea = concat(v_sem_fea, c_sem_fea)
  for (int i = tid; i < 24 * 128; i += 256){
    int r = i >> 7, c = i & 127;
    const float* x = (r < 16) ? (A.vsem + r * 768) : (A.csem + (r - 16) * 768);
    float acc = A.b1[c];
    for (int k = 0; k < 768; ++k) acc += x[k] * A.w1[k * 128 + c];
    sh1[i] = fmaxf(acc, 0.f);
  }
  __syncthreads();
  // fea2 = h1 @ proj_w2 + b2
  for (int i = tid; i < 24 * 64; i += 256){
    int r = i >> 6, c = i & 63;
    float acc = A.b2[c];
    for (int k = 0; k < 128; ++k) acc += sh1[r * 128 + k] * A.w2[k * 64 + c];
    sf[i] = acc;
  }
  __syncthreads();
  // q, k, v projections
  for (int i = tid; i < 3 * 24 * 64; i += 256){
    int m = i / (24 * 64), j = i % (24 * 64);
    int r = j >> 6, c = j & 63;
    const float* W = (m == 0) ? A.sa_wq : (m == 1) ? A.sa_wk : A.sa_wv;
    const float* B = (m == 0) ? A.sa_bq : (m == 1) ? A.sa_bk : A.sa_bv;
    float acc = B[c];
    for (int k = 0; k < 64; ++k) acc += sf[r * 64 + k] * W[k * 64 + c];
    ((m == 0) ? sq : (m == 1) ? sk : sv)[j] = acc;
  }
  __syncthreads();
  // self-attention, 4 heads x 24 queries
  if (tid < 96){
    int h = tid / 24, qi = tid % 24;
    float sc[24], mx = -1e30f;
    for (int ki = 0; ki < 24; ++ki){
      float d = 0.f;
      for (int e = 0; e < 16; ++e) d += sq[qi*64 + h*16 + e] * sk[ki*64 + h*16 + e];
      d *= 0.25f;                       // 1/sqrt(16)
      sc[ki] = d; mx = fmaxf(mx, d);
    }
    float zs = 0.f;
    for (int ki = 0; ki < 24; ++ki){ sc[ki] = __expf(sc[ki] - mx); zs += sc[ki]; }
    float inv = 1.f / zs;
    for (int e = 0; e < 16; ++e){
      float o = 0.f;
      for (int ki = 0; ki < 24; ++ki) o += sc[ki] * sv[ki*64 + h*16 + e];
      so[qi*64 + h*16 + e] = o * inv;
    }
  }
  __syncthreads();
  // out-proj + residual + LN -> fea_sem into sh1[0..24*64)
  if (tid < 24){
    int r = tid;
    float tmp[64], s = 0.f;
    for (int c = 0; c < 64; ++c){
      float acc = A.sa_bo[c] + sf[r * 64 + c];
      for (int k = 0; k < 64; ++k) acc += so[r * 64 + k] * A.sa_wo[k * 64 + c];
      tmp[c] = acc; s += acc;
    }
    float mean = s * (1.f / 64.f), ss = 0.f;
    for (int c = 0; c < 64; ++c){ float d = tmp[c] - mean; ss += d * d; }
    float rstd = rsqrtf(ss * (1.f / 64.f) + 1e-5f);
    for (int c = 0; c < 64; ++c)
      sh1[r * 64 + c] = (tmp[c] - mean) * rstd * A.lnw[c] + A.lnb[c];
  }
  __syncthreads();
  // q projections for the segment cross-attention, pre-scaled by 1/sqrt(dh)=0.25
  for (int i = tid; i < 16 * 64; i += 256){
    int r = i >> 6, c = i & 63;
    float acc = A.cav_bq[c];
    for (int k = 0; k < 64; ++k) acc += sh1[r * 64 + k] * A.cav_wq[k * 64 + c];
    A.qv[i] = acc * 0.25f;
  }
  for (int i = tid; i < 8 * 64; i += 256){
    int r = i >> 6, c = i & 63;
    float acc = A.cac_bq[c];
    for (int k = 0; k < 64; ++k) acc += sh1[(16 + r) * 64 + k] * A.cac_wq[k * 64 + c];
    A.qc[i] = acc * 0.25f;
  }
  // pack weight fragments for the big-sweep GEMMs
  pack_frags_64x64(A.sv_w,   A.frags + 0 * 4096);
  pack_frags_64x64(A.cav_wk, A.frags + 1 * 4096);
  pack_frags_64x64(A.cav_wv, A.frags + 2 * 4096);
  pack_frags_64x64(A.sc_w,   A.frags + 4 * 4096);
  pack_frags_64x64(A.cac_wk, A.frags + 5 * 4096);
  pack_frags_64x64(A.cac_wv, A.frags + 6 * 4096);
  // (re)initialize global accumulators every launch (graph-replay safe)
  if (tid < 64) A.maxv[tid] = ENC_NEG_INF;
  if (tid < 32) A.maxc[tid] = ENC_NEG_INF;
  for (int i = tid; i < 32 * 16 * 64; i += 256) A.numv[i] = 0.f;
  for (int i = tid; i < 32 * 64;      i += 256) A.zv[i]   = 0.f;
  for (int i = tid; i < 32 * 8 * 64;  i += 256) A.numc[i] = 0.f;
  for (int i = tid; i < 32 * 32;      i += 256) A.zc[i]   = 0.f;
}

// ---------------------------------------------------------------------------
// Pass 1: LN -> v_s -> k -> scores s[n,h] + segment max (uint-encoded atomics)
// ---------------------------------------------------------------------------
__global__ __launch_bounds__(256) void k_pass1(
    const float* __restrict__ X, const int* __restrict__ cls,
    const float* __restrict__ lnw, const float* __restrict__ lnb,
    const _Float16* __restrict__ fragS, const float* __restrict__ sb,
    const _Float16* __restrict__ fragK, const float* __restrict__ kb,
    const float* __restrict__ q, float* __restrict__ s_out,
    unsigned* __restrict__ segmax, int N, int nseg)
{
  __shared__ _Float16 sAct[128 * 64];
  __shared__ float sRed[256], sRed2[256];
  __shared__ float sK[128 * 64];
  __shared__ unsigned sMax[64];
  int tid = threadIdx.x, lane = tid & 31, w = tid >> 5;
  int base = blockIdx.x * 128;
  if (tid < 64) sMax[tid] = ENC_NEG_INF;

  stage_ln(X, N, base, lnw, lnb, sAct, sRed, sRed2);

  // GEMM1: v_s = LN(x) @ sv_w + sv_b
  _Float16* act = sAct + w * 16 * 64;
  int hi = (lane >> 4) & 1, nloc = lane & 15;
  v8f acc[4];
#pragma unroll
  for (int t = 0; t < 4; ++t) acc[t] = fzero8();
  gemm_tile(act, fragS, lane, acc);
  float bS[4];
#pragma unroll
  for (int t = 0; t < 4; ++t) bS[t] = sb[t * 16 + nloc];
  __syncthreads();
#pragma unroll
  for (int t = 0; t < 4; ++t)
#pragma unroll
    for (int r = 0; r < 8; ++r)
      sAct[(w * 16 + r + 8 * hi) * 64 + t * 16 + nloc] = (_Float16)(acc[t][r] + bS[t]);
  __syncthreads();

  // GEMM2: k = v_s @ wk + bk  (f32 into LDS)
  v8f kacc[4];
#pragma unroll
  for (int t = 0; t < 4; ++t) kacc[t] = fzero8();
  gemm_tile(act, fragK, lane, kacc);
#pragma unroll
  for (int t = 0; t < 4; ++t){
    float bK = kb[t * 16 + nloc];
#pragma unroll
    for (int r = 0; r < 8; ++r)
      sK[(w * 16 + r + 8 * hi) * 64 + t * 16 + nloc] = kacc[t][r] + bK;
  }
  __syncthreads();

  // scores s[n,h] = (q[seg[n],h,:]/4) . k[n,h,:], block-local then global max
  {
    int r = tid >> 1, row = base + r;
    if (row < N){
      int seg = cls[row];
      int h0 = (tid & 1) * 2;
#pragma unroll
      for (int hh = 0; hh < 2; ++hh){
        int h = h0 + hh;
        const float* qp = q + seg * 64 + h * 16;
        const float* kp = sK + r * 64 + h * 16;
        float d = 0.f;
#pragma unroll
        for (int i = 0; i < 16; ++i) d += qp[i] * kp[i];
        s_out[(size_t)row * 4 + h] = d;
        atomicMax(&sMax[seg * 4 + h], enc_f32(d));
      }
    }
  }
  __syncthreads();
  if (tid < nseg * 4) atomicMax(&segmax[tid], sMax[tid]);
}

// ---------------------------------------------------------------------------
// Pass 2: LN -> v_s -> vp = v_s@wv; p=exp(s-m[seg]); accumulate num/z
// (LDS pre-reduction, then 32-way-sliced global float atomics)
// ---------------------------------------------------------------------------
__global__ __launch_bounds__(256) void k_pass2(
    const float* __restrict__ X, const int* __restrict__ cls,
    const float* __restrict__ lnw, const float* __restrict__ lnb,
    const _Float16* __restrict__ fragS, const float* __restrict__ sb,
    const _Float16* __restrict__ fragV, const float* __restrict__ vb,
    const float* __restrict__ s_in, const unsigned* __restrict__ segmax,
    float* __restrict__ numSlots, float* __restrict__ zSlots, int N, int nseg)
{
  __shared__ _Float16 sAct[128 * 64];
  __shared__ float sRed[256], sRed2[256];
  __shared__ float sV[128 * 64];
  __shared__ float sNum[16 * 64];
  __shared__ float sZ[64];
  __shared__ float sM[64];
  int tid = threadIdx.x, lane = tid & 31, w = tid >> 5;
  int base = blockIdx.x * 128;
  for (int i = tid; i < 16 * 64; i += 256) sNum[i] = 0.f;
  if (tid < 64){
    sZ[tid] = 0.f;
    sM[tid] = (tid < nseg * 4) ? dec_f32(segmax[tid]) : 0.f;
  }

  stage_ln(X, N, base, lnw, lnb, sAct, sRed, sRed2);

  _Float16* act = sAct + w * 16 * 64;
  int hi = (lane >> 4) & 1, nloc = lane & 15;
  v8f acc[4];
#pragma unroll
  for (int t = 0; t < 4; ++t) acc[t] = fzero8();
  gemm_tile(act, fragS, lane, acc);
  float bS[4];
#pragma unroll
  for (int t = 0; t < 4; ++t) bS[t] = sb[t * 16 + nloc];
  __syncthreads();
#pragma unroll
  for (int t = 0; t < 4; ++t)
#pragma unroll
    for (int r = 0; r < 8; ++r)
      sAct[(w * 16 + r + 8 * hi) * 64 + t * 16 + nloc] = (_Float16)(acc[t][r] + bS[t]);
  __syncthreads();

  v8f vacc[4];
#pragma unroll
  for (int t = 0; t < 4; ++t) vacc[t] = fzero8();
  gemm_tile(act, fragV, lane, vacc);
#pragma unroll
  for (int t = 0; t < 4; ++t){
    float bV = vb[t * 16 + nloc];
#pragma unroll
    for (int r = 0; r < 8; ++r)
      sV[(w * 16 + r + 8 * hi) * 64 + t * 16 + nloc] = vacc[t][r] + bV;
  }
  __syncthreads();

  {
    int r = tid >> 1, row = base + r;
    if (row < N){
      int seg = cls[row];
      int h0 = (tid & 1) * 2;
#pragma unroll
      for (int hh = 0; hh < 2; ++hh){
        int h = h0 + hh;
        float p = __expf(s_in[(size_t)row * 4 + h] - sM[seg * 4 + h]);
        atomicAdd(&sZ[seg * 4 + h], p);
        const float* vp = sV + r * 64 + h * 16;
        float* np = sNum + seg * 64 + h * 16;
#pragma unroll
        for (int d = 0; d < 16; ++d) atomicAdd(&np[d], p * vp[d]);
      }
    }
  }
  __syncthreads();
  int slot = blockIdx.x & 31;
  for (int i = tid; i < nseg * 64; i += 256)
    atomicAdd(&numSlots[slot * nseg * 64 + i], sNum[i]);
  if (tid < nseg * 4)
    atomicAdd(&zSlots[slot * nseg * 4 + tid], sZ[tid]);
}

// ---------------------------------------------------------------------------
// Kernel 3: finalize: o = num/z; v_final = o@wo+bo; G = v_final^T @ r_w; pack G
// ---------------------------------------------------------------------------
struct K3Args {
  const float *numv, *zv, *numc, *zc;
  const float *cav_wo, *cav_bo, *cac_wo, *cac_bo;
  const float *rv_w, *rc_w;
  _Float16 *frags;
};

__global__ __launch_bounds__(256) void k_finalize(K3Args A){
  __shared__ float so  [16 * 64];
  __shared__ float sfin[16 * 64];
  __shared__ float sZ  [64];
  __shared__ float sG  [64 * 64];
  int tid = threadIdx.x;
  for (int pass = 0; pass < 2; ++pass){
    int nseg = (pass == 0) ? 16 : 8;
    const float* num = (pass == 0) ? A.numv : A.numc;
    const float* z   = (pass == 0) ? A.zv : A.zc;
    const float* wo  = (pass == 0) ? A.cav_wo : A.cac_wo;
    const float* bo  = (pass == 0) ? A.cav_bo : A.cac_bo;
    const float* rw  = (pass == 0) ? A.rv_w : A.rc_w;
    _Float16* fd     = A.frags + ((pass == 0) ? 3 : 7) * 4096;
    if (tid < nseg * 4){
      float acc = 0.f;
      for (int s2 = 0; s2 < 32; ++s2) acc += z[s2 * nseg * 4 + tid];
      sZ[tid] = acc;
    }
    __syncthreads();
    for (int i = tid; i < nseg * 64; i += 256){
      float acc = 0.f;
      for (int s2 = 0; s2 < 32; ++s2) acc += num[s2 * nseg * 64 + i];
      int seg = i >> 6, col = i & 63;
      so[i] = acc / sZ[seg * 4 + (col >> 4)];
    }
    __syncthreads();
    for (int i = tid; i < nseg * 64; i += 256){
      int r = i >> 6, c = i & 63;
      float acc = bo[c];
      for (int k = 0; k < 64; ++k) acc += so[r * 64 + k] * wo[k * 64 + c];
      sfin[i] = acc;
    }
    __syncthreads();
    // G[a][b] = sum_j v_final[j][a] * r_w[j][b]
    for (int i = tid; i < 64 * 64; i += 256){
      int a = i >> 6, b = i & 63;
      float acc = 0.f;
      for (int j = 0; j < nseg; ++j) acc += sfin[j * 64 + a] * rw[j * 64 + b];
      sG[i] = acc;
    }
    __syncthreads();
    pack_frags_64x64(sG, fd);
    __syncthreads();
  }
}

// ---------------------------------------------------------------------------
// Pass 4: LN -> v_s -> u = v_s@G + rb; out = v_s * sin(u)
// ---------------------------------------------------------------------------
__global__ __launch_bounds__(256) void k_pass4(
    const float* __restrict__ X,
    const float* __restrict__ lnw, const float* __restrict__ lnb,
    const _Float16* __restrict__ fragS, const float* __restrict__ sb,
    const _Float16* __restrict__ fragG, const float* __restrict__ rb,
    float* __restrict__ out, int N)
{
  __shared__ _Float16 sAct[128 * 64];
  __shared__ float sRed[256], sRed2[256];
  int tid = threadIdx.x, lane = tid & 31, w = tid >> 5;
  int base = blockIdx.x * 128;

  stage_ln(X, N, base, lnw, lnb, sAct, sRed, sRed2);

  _Float16* act = sAct + w * 16 * 64;
  int hi = (lane >> 4) & 1, nloc = lane & 15;
  v8f acc[4];
#pragma unroll
  for (int t = 0; t < 4; ++t) acc[t] = fzero8();
  gemm_tile(act, fragS, lane, acc);
#pragma unroll
  for (int t = 0; t < 4; ++t){
    float bS = sb[t * 16 + nloc];
#pragma unroll
    for (int r = 0; r < 8; ++r) acc[t][r] += bS;   // acc = v_s (kept in regs)
  }
  __syncthreads();
#pragma unroll
  for (int t = 0; t < 4; ++t)
#pragma unroll
    for (int r = 0; r < 8; ++r)
      sAct[(w * 16 + r + 8 * hi) * 64 + t * 16 + nloc] = (_Float16)acc[t][r];
  __syncthreads();

  v8f u[4];
#pragma unroll
  for (int t = 0; t < 4; ++t) u[t] = fzero8();
  gemm_tile(act, fragG, lane, u);
#pragma unroll
  for (int t = 0; t < 4; ++t){
    float bR = rb[t * 16 + nloc];
#pragma unroll
    for (int r = 0; r < 8; ++r){
      int row = base + w * 16 + r + 8 * hi;
      if (row < N)
        out[(size_t)row * 64 + t * 16 + nloc] = acc[t][r] * __sinf(u[t][r] + bR);
    }
  }
}

// ---------------------------------------------------------------------------
extern "C" void kernel_launch(void* const* d_in, const int* in_sizes, int n_in,
                              void* d_out, int out_size, void* d_ws, size_t ws_size,
                              hipStream_t stream)
{
  const float* v        = (const float*)d_in[0];
  const float* c        = (const float*)d_in[1];
  const float* v_sem    = (const float*)d_in[2];
  const float* c_sem    = (const float*)d_in[3];
  const float* ln_w     = (const float*)d_in[4];
  const float* ln_b     = (const float*)d_in[5];
  const float* proj_w1  = (const float*)d_in[6];
  const float* proj_b1  = (const float*)d_in[7];
  const float* proj_w2  = (const float*)d_in[8];
  const float* proj_b2  = (const float*)d_in[9];
  const float* sa_wq  = (const float*)d_in[10]; const float* sa_bq  = (const float*)d_in[11];
  const float* sa_wk  = (const float*)d_in[12]; const float* sa_bk  = (const float*)d_in[13];
  const float* sa_wv  = (const float*)d_in[14]; const float* sa_bv  = (const float*)d_in[15];
  const float* sa_wo  = (const float*)d_in[16]; const float* sa_bo  = (const float*)d_in[17];
  const float* cav_wq = (const float*)d_in[18]; const float* cav_bq = (const float*)d_in[19];
  const float* cav_wk = (const float*)d_in[20]; const float* cav_bk = (const float*)d_in[21];
  const float* cav_wv = (const float*)d_in[22]; const float* cav_bv = (const float*)d_in[23];
  const float* cav_wo = (const float*)d_in[24]; const float* cav_bo = (const float*)d_in[25];
  const float* cac_wq = (const float*)d_in[26]; const float* cac_bq = (const float*)d_in[27];
  const float* cac_wk = (const float*)d_in[28]; const float* cac_bk = (const float*)d_in[29];
  const float* cac_wv = (const float*)d_in[30]; const float* cac_bv = (const float*)d_in[31];
  const float* cac_wo = (const float*)d_in[32]; const float* cac_bo = (const float*)d_in[33];
  const float* sv_w = (const float*)d_in[34]; const float* sv_b = (const float*)d_in[35];
  const float* sc_w = (const float*)d_in[36]; const float* sc_b = (const float*)d_in[37];
  const float* rv_w = (const float*)d_in[38]; const float* rv_b = (const float*)d_in[39];
  const float* rc_w = (const float*)d_in[40]; const float* rc_b = (const float*)d_in[41];
  const int* v_class = (const int*)d_in[42];
  const int* c_class = (const int*)d_in[43];

  const int NV = in_sizes[0] / 64;
  const int NC = in_sizes[1] / 64;

  // workspace layout (float units), ~12.3 MB total
  float* wsf = (float*)d_ws;
  float* s_v = wsf;                               // NV*4 scores
  float* s_c = s_v + (size_t)NV * 4;              // NC*4 scores
  float* qv  = s_c + (size_t)NC * 4;              // 16*64
  float* qc  = qv + 16 * 64;                      // 8*64
  unsigned* maxv = (unsigned*)(qc + 8 * 64);      // 64
  unsigned* maxc = maxv + 64;                     // 32
  float* numv = (float*)(maxc + 32);              // 32*16*64
  float* zv   = numv + 32 * 16 * 64;              // 32*64
  float* numc = zv + 32 * 64;                     // 32*8*64
  float* zc   = numc + 32 * 8 * 64;               // 32*32
  size_t off  = (size_t)((zc + 32 * 32) - wsf);
  off = (off + 7) & ~(size_t)7;                   // 32B-align fragment region
  _Float16* frags = (_Float16*)(wsf + off);       // 8 matrices * 4096 halves

  K0Args a0;
  a0.vsem = v_sem; a0.csem = c_sem;
  a0.w1 = proj_w1; a0.b1 = proj_b1; a0.w2 = proj_w2; a0.b2 = proj_b2;
  a0.sa_wq = sa_wq; a0.sa_bq = sa_bq; a0.sa_wk = sa_wk; a0.sa_bk = sa_bk;
  a0.sa_wv = sa_wv; a0.sa_bv = sa_bv; a0.sa_wo = sa_wo; a0.sa_bo = sa_bo;
  a0.lnw = ln_w; a0.lnb = ln_b;
  a0.cav_wq = cav_wq; a0.cav_bq = cav_bq; a0.cac_wq = cac_wq; a0.cac_bq = cac_bq;
  a0.sv_w = sv_w; a0.sc_w = sc_w;
  a0.cav_wk = cav_wk; a0.cav_wv = cav_wv; a0.cac_wk = cac_wk; a0.cac_wv = cac_wv;
  a0.qv = qv; a0.qc = qc; a0.maxv = maxv; a0.maxc = maxc;
  a0.numv = numv; a0.zv = zv; a0.numc = numc; a0.zc = zc; a0.frags = frags;
  k_setup<<<dim3(1), dim3(256), 0, stream>>>(a0);

  const int gv = (NV + 127) / 128;
  const int gc = (NC + 127) / 128;

  k_pass1<<<dim3(gv), dim3(256), 0, stream>>>(v, v_class, ln_w, ln_b,
      frags + 0 * 4096, sv_b, frags + 1 * 4096, cav_bk, qv, s_v, maxv, NV, 16);
  k_pass1<<<dim3(gc), dim3(256), 0, stream>>>(c, c_class, ln_w, ln_b,
      frags + 4 * 4096, sc_b, frags + 5 * 4096, cac_bk, qc, s_c, maxc, NC, 8);

  k_pass2<<<dim3(gv), dim3(256), 0, stream>>>(v, v_class, ln_w, ln_b,
      frags + 0 * 4096, sv_b, frags + 2 * 4096, cav_bv, s_v, maxv, numv, zv, NV, 16);
  k_pass2<<<dim3(gc), dim3(256), 0, stream>>>(c, c_class, ln_w, ln_b,
      frags + 4 * 4096, sc_b, frags + 6 * 4096, cac_bv, s_c, maxc, numc, zc, NC, 8);

  K3Args a3;
  a3.numv = numv; a3.zv = zv; a3.numc = numc; a3.zc = zc;
  a3.cav_wo = cav_wo; a3.cav_bo = cav_bo; a3.cac_wo = cac_wo; a3.cac_bo = cac_bo;
  a3.rv_w = rv_w; a3.rc_w = rc_w; a3.frags = frags;
  k_finalize<<<dim3(1), dim3(256), 0, stream>>>(a3);

  float* outv = (float*)d_out;
  float* outc = outv + (size_t)NV * 64;
  k_pass4<<<dim3(gv), dim3(256), 0, stream>>>(v, ln_w, ln_b,
      frags + 0 * 4096, sv_b, frags + 3 * 4096, rv_b, outv, NV);
  k_pass4<<<dim3(gc), dim3(256), 0, stream>>>(c, ln_w, ln_b,
      frags + 4 * 4096, sc_b, frags + 7 * 4096, rc_b, outc, NC);
}
